// Weather_forcast_20418274526007
// MI455X (gfx1250) — matrix operand order repaired and checked by reading, not compiled, not run on hardware
//
#include <hip/hip_runtime.h>
#include <hip/hip_bf16.h>

// ---------------------------------------------------------------------------
// HMM forward pass on gfx1250.
//   log-space recurrence turned into bf16 WMMA GEMMs:
//     A[b,k]   = exp(alpha[b,k] - rowmax(alpha[b,:]))        (bf16, in LDS)
//     B[k,i]   = exp(log_softmax(T, axis=0))[i,k]            (bf16, ws, i-major)
//     new[b,i] = rowmax + log( (A x B^T)[b,i] ) + logE[i, sym[b,t+1]]
// ---------------------------------------------------------------------------

#define N_STATES 512
#define M_SYMS   256
#define BATCH    256
#define TILE_B   16

typedef __attribute__((ext_vector_type(16))) __bf16 v16bf;
typedef __attribute__((ext_vector_type(8)))  float  v8f;
typedef unsigned int u32x4 __attribute__((ext_vector_type(4)));

union FragBF { v16bf v; u32x4 q[2]; };

__device__ __forceinline__ unsigned short f2bf(float f) {
    unsigned int u = __float_as_uint(f);
    unsigned int r = u + 0x7FFFu + ((u >> 16) & 1u);   // round-to-nearest-even
    return (unsigned short)(r >> 16);
}

// ---- transition column log-sum-exp:  lse_col[k] = LSE_i U[i,k] ---------------
__global__ void hmm_col_lse(const float* __restrict__ U, float* __restrict__ lse_col) {
    __shared__ float red[256];
    const int k = blockIdx.x, tid = threadIdx.x;
    float v0 = U[tid * N_STATES + k];
    float v1 = U[(tid + 256) * N_STATES + k];
    red[tid] = fmaxf(v0, v1);
    __syncthreads();
    for (int s = 128; s > 0; s >>= 1) {
        if (tid < s) red[tid] = fmaxf(red[tid], red[tid + s]);
        __syncthreads();
    }
    float m = red[0];
    __syncthreads();
    red[tid] = __expf(v0 - m) + __expf(v1 - m);
    __syncthreads();
    for (int s = 128; s > 0; s >>= 1) {
        if (tid < s) red[tid] += red[tid + s];
        __syncthreads();
    }
    if (tid == 0) lse_col[k] = m + __logf(red[0]);
}

// ---- TBt[i][k] = bf16( exp(U[i,k] - lse_col[k]) ), i-major -------------------
__global__ void hmm_build_tbt(const float* __restrict__ U,
                              const float* __restrict__ lse_col,
                              unsigned short* __restrict__ TBt) {
    const int i = blockIdx.x, tid = threadIdx.x;
    for (int k = tid; k < N_STATES; k += 256)
        TBt[i * N_STATES + k] = f2bf(__expf(U[i * N_STATES + k] - lse_col[k]));
}

// ---- emission row log-softmax: logE[n, :] = U[n,:] - LSE_m U[n,m] ------------
__global__ void hmm_emis(const float* __restrict__ U, float* __restrict__ logE) {
    __shared__ float red[256];
    const int n = blockIdx.x, tid = threadIdx.x;
    float v = U[n * M_SYMS + tid];
    red[tid] = v;
    __syncthreads();
    for (int s = 128; s > 0; s >>= 1) {
        if (tid < s) red[tid] = fmaxf(red[tid], red[tid + s]);
        __syncthreads();
    }
    float m = red[0];
    __syncthreads();
    red[tid] = __expf(v - m);
    __syncthreads();
    for (int s = 128; s > 0; s >>= 1) {
        if (tid < s) red[tid] += red[tid + s];
        __syncthreads();
    }
    logE[n * M_SYMS + tid] = v - m - __logf(red[0]);
}

// ---- prior log-softmax -------------------------------------------------------
__global__ void hmm_prior(const float* __restrict__ U, float* __restrict__ logP) {
    __shared__ float red[512];
    const int tid = threadIdx.x;
    float v = U[tid];
    red[tid] = v;
    __syncthreads();
    for (int s = 256; s > 0; s >>= 1) {
        if (tid < s) red[tid] = fmaxf(red[tid], red[tid + s]);
        __syncthreads();
    }
    float m = red[0];
    __syncthreads();
    red[tid] = __expf(v - m);
    __syncthreads();
    for (int s = 256; s > 0; s >>= 1) {
        if (tid < s) red[tid] += red[tid + s];
        __syncthreads();
    }
    logP[tid] = v - m - __logf(red[0]);
}

// ---- main forward kernel: one workgroup = 16 batch rows, 8 waves x 64 cols ---
__global__ __launch_bounds__(256) void hmm_forward(
        const int*            __restrict__ batch,   // [BATCH][M_SYMS]
        const float*          __restrict__ logE,    // [N][M]
        const float*          __restrict__ logP,    // [N]
        const unsigned short* __restrict__ TBt,     // [N(i)][N(k)] bf16
        float*                __restrict__ out) {   // [BATCH][M_SYMS]

    __shared__ float          alpha_s[TILE_B][N_STATES + 1];   // f32 alpha tile
    __shared__ unsigned short aexp_s[TILE_B][N_STATES + 8];    // bf16 exp(alpha-m)
    __shared__ float          part_s[TILE_B][16];
    __shared__ float          rowm_s[TILE_B];
    __shared__ int            sym_s[TILE_B];

    const int tid  = threadIdx.x;
    const int b0   = blockIdx.x * TILE_B;
    const int lane = tid & 31;
    const int wave = tid >> 5;
    const int hi   = lane >> 4;     // half-wave select
    const int nlow = lane & 15;
    const int r    = tid >> 4;      // row for stats phase (0..15)
    const int j    = tid & 15;      // column group for stats phase

    // ---- alpha_0 = logE[n, sym(b,0)] + logP[n]
    if (tid < TILE_B) sym_s[tid] = batch[(b0 + tid) * M_SYMS];
    __syncthreads();
    for (int idx = tid; idx < TILE_B * N_STATES; idx += 256) {
        int b = idx >> 9, n = idx & (N_STATES - 1);
        alpha_s[b][n] = logE[n * M_SYMS + sym_s[b]] + logP[n];
    }
    __syncthreads();

    for (int t = 0; t < M_SYMS; ++t) {
        // ---- row max ----
        float pmax = -3.4e38f;
        for (int k = j; k < N_STATES; k += 16) pmax = fmaxf(pmax, alpha_s[r][k]);
        part_s[r][j] = pmax;
        __syncthreads();
        if (j == 0) {
            float m = part_s[r][0];
            for (int jj = 1; jj < 16; ++jj) m = fmaxf(m, part_s[r][jj]);
            rowm_s[r] = m;
        }
        __syncthreads();

        // ---- exp(alpha-m) -> bf16 LDS, and row sum for the per-step output ----
        const float m = rowm_s[r];
        float psum = 0.f;
        for (int k = j; k < N_STATES; k += 16) {
            float e = __expf(alpha_s[r][k] - m);
            psum += e;
            aexp_s[r][k] = f2bf(e);
        }
        part_s[r][j] = psum;
        __syncthreads();
        if (j == 0) {
            float s = 0.f;
            for (int jj = 0; jj < 16; ++jj) s += part_s[r][jj];
            out[(b0 + r) * M_SYMS + t] = m + __logf(s);   // LSE_n alpha_t[b,n]
        }
        if (t == M_SYMS - 1) break;

        if (tid < TILE_B) sym_s[tid] = batch[(b0 + tid) * M_SYMS + t + 1];
        __syncthreads();

        // Opaque (always-zero) offset, renewed each t-iteration: defeats LICM so
        // the 128 B-fragment loads stay inside the recurrence loop (no 512-VGPR
        // hoist + scratch spill), while keeping the pointer rooted at the global
        // kernel arg so they lower as global_load_b128 (LOADcnt only, no DScnt).
        size_t b_ofs = 0;
        asm volatile("" : "+s"(b_ofs));
        const unsigned short* TBt_it = TBt + b_ofs;

        // ---- GEMM: new_exp[b, n0..n0+63] = A(16x512) x B^T ----
        const int n0 = wave * 64;
        v8f z;
        #pragma unroll
        for (int q = 0; q < 8; ++q) z[q] = 0.f;
        v8f acc[4] = {z, z, z, z};

        for (int kc = 0; kc < 16; ++kc) {
            const int kb = kc * 32;
            // A fragment (16-bit A 16x32 lane layout): lane<16 -> K 0..7 / 16..23
            FragBF fa;
            const unsigned short* ap = &aexp_s[nlow][kb + hi * 8];
            fa.q[0] = *(const u32x4*)(ap);
            fa.q[1] = *(const u32x4*)(ap + 16);

            const int kbB = kb + hi * 8;    // B: lane<16 -> K 0..7 / 16..23, else 8..15 / 24..31
            #pragma unroll
            for (int nt = 0; nt < 4; ++nt) {
                const unsigned short* bp =
                    TBt_it + (size_t)(n0 + nt * 16 + nlow) * N_STATES + kbB;
                FragBF fb;
                fb.q[0] = *(const u32x4*)(bp);
                fb.q[1] = *(const u32x4*)(bp + 16);
                if (kc < 15) __builtin_prefetch(bp + 32, 0, 3);  // near-scope RT
                acc[nt] = __builtin_amdgcn_wmma_f32_16x16x32_bf16(
                    false, fa.v, false, fb.v, (short)0, acc[nt], false, false);
            }
        }

        // ---- epilogue: alpha_{t+1}[b,n] = m[b] + log(acc) + logE[n, sym(b,t+1)]
        // C layout: vgpr v, lane l: row = v + 8*(l>=16), col = l&15
        #pragma unroll
        for (int nt = 0; nt < 4; ++nt) {
            const int n = n0 + nt * 16 + nlow;
            const float* lEn = logE + (size_t)n * M_SYMS;
            #pragma unroll
            for (int v = 0; v < 8; ++v) {
                const int bl = hi * 8 + v;
                alpha_s[bl][n] = rowm_s[bl] + __logf(acc[nt][v]) + lEn[sym_s[bl]];
            }
        }
        __syncthreads();
    }
}

extern "C" void kernel_launch(void* const* d_in, const int* in_sizes, int n_in,
                              void* d_out, int out_size, void* d_ws, size_t ws_size,
                              hipStream_t stream) {
    (void)in_sizes; (void)n_in; (void)out_size; (void)ws_size;
    const int*   batch = (const int*)d_in[0];
    const float* U_E   = (const float*)d_in[1];
    const float* U_T   = (const float*)d_in[2];
    const float* U_P   = (const float*)d_in[3];
    float*       out   = (float*)d_out;

    char* ws = (char*)d_ws;
    unsigned short* TBt     = (unsigned short*)(ws);                    // 512 KB
    float*          logE    = (float*)(ws + (size_t)512 * 1024);        // 512 KB
    float*          logP    = (float*)(ws + (size_t)1024 * 1024);       //   2 KB
    float*          lse_col = (float*)(ws + (size_t)1024 * 1024 + 4096);

    hmm_col_lse  <<<N_STATES, 256, 0, stream>>>(U_T, lse_col);
    hmm_build_tbt<<<N_STATES, 256, 0, stream>>>(U_T, lse_col, TBt);
    hmm_emis     <<<N_STATES, 256, 0, stream>>>(U_E, logE);
    hmm_prior    <<<1,        512, 0, stream>>>(U_P, logP);
    hmm_forward  <<<BATCH / TILE_B, 256, 0, stream>>>(batch, logE, logP, TBt, out);
}
